// MiniGPT_55551107007174
// MI455X (gfx1250) — matrix operand order repaired and checked by reading, not compiled, not run on hardware
//
#include <hip/hip_runtime.h>
#include <math.h>

// ---------------------------------------------------------------------------
// MiniGPT forward for MI455X (gfx1250): bf16 WMMA GEMMs + fused flash attn.
// Residual stream fp32; all matmuls via v_wmma_f32_16x16x32_bf16.
// GEMM A tiles moved by the Tensor Data Mover (tensor_load_to_lds, TENSORcnt);
// attention K tiles staged with global_load_async_to_lds_b128 (ASYNCcnt).
// ---------------------------------------------------------------------------

typedef __attribute__((ext_vector_type(16))) __bf16       v16bf;
typedef __attribute__((ext_vector_type(8)))  float        v8f;
typedef __attribute__((ext_vector_type(4)))  unsigned int u32x4;
typedef __attribute__((ext_vector_type(8)))  unsigned int u32x8;
typedef __attribute__((ext_vector_type(4)))  float        f32x4;

union Frag {            // 16 bf16 per lane = 8 VGPRs (wave32 WMMA operand)
    v16bf v;
    u32x4 q[2];
    unsigned short u[16];
};

__device__ __forceinline__ unsigned short f2bf(float f) {
    unsigned int u = __float_as_uint(f);
    unsigned int r = u + 0x7FFFu + ((u >> 16) & 1u);   // round-to-nearest-even
    return (unsigned short)(r >> 16);
}

__device__ __forceinline__ v8f zero8() {
    v8f z;
#pragma unroll
    for (int i = 0; i < 8; ++i) z[i] = 0.0f;
    return z;
}

__device__ __forceinline__ f32x4 fzero4() {
    f32x4 f;
    f.x = 0.f; f.y = 0.f; f.z = 0.f; f.w = 0.f;
    return f;
}

// LDS byte offset of a __shared__ object (low 32 bits of the LDS aperture addr)
__device__ __forceinline__ unsigned lds_off(const void* p) {
    return (unsigned)(uintptr_t)p;
}

// Async global->LDS copy, 16B per lane, GVS addressing (SADDR base + 32b voff).
__device__ __forceinline__ void async_b128(unsigned lds_byte, unsigned goff_byte,
                                           const void* base) {
    asm volatile("global_load_async_to_lds_b128 %0, %1, %2"
                 :: "v"(lds_byte), "v"(goff_byte), "s"(base)
                 : "memory");
}

__device__ __forceinline__ void wait_async0() {
    asm volatile("s_wait_asynccnt 0" ::: "memory");
}

// Tensor Data Mover: 2D tile load, global -> LDS (D# per CDNA5 ISA).
// data_size = 2 bytes (bf16). All arguments must be wave-uniform (SGPRs).
__device__ __forceinline__ void tdm_load_2d(
    unsigned lds_byte,          // LDS destination byte address
    const void* gtile,          // global address of tile start
    unsigned tensor_dim0,       // tensor row length (elements)
    unsigned tensor_dim1,       // tensor rows
    unsigned tile_dim0,         // tile row length (elements)
    unsigned tile_dim1,         // tile rows
    unsigned stride0)           // row stride (elements)
{
    unsigned long long ga = (unsigned long long)(uintptr_t)gtile;
    u32x4 g0;
    g0[0] = 1u;                                            // count=1 (valid), user
    g0[1] = lds_byte;                                      // lds_addr
    g0[2] = (unsigned)(ga & 0xFFFFFFFFu);                  // global_addr[31:0]
    g0[3] = (unsigned)((ga >> 32) & 0x01FFFFFFu) | (2u << 30);  // [56:32] | type=2
    u32x8 g1;
    g1[0] = 1u << 16;                                      // data_size=1 -> 2 bytes
    g1[1] = (tensor_dim0 & 0xFFFFu) << 16;                 // dim0[15:0] at bits 63:48
    g1[2] = (tensor_dim0 >> 16) | ((tensor_dim1 & 0xFFFFu) << 16);
    g1[3] = (tensor_dim1 >> 16) | (tile_dim0 << 16);       // tile_dim0 at 127:112
    g1[4] = tile_dim1 & 0xFFFFu;                           // tile_dim1; tile_dim2=0
    g1[5] = stride0;                                       // dim0_stride[31:0]
    g1[6] = 0u;                                            // stride hi, dim1_stride lo
    g1[7] = 0u;
    asm volatile("tensor_load_to_lds %0, %1, null, null"
                 :: "s"(g0), "s"(g1) : "memory");
}

// Model constants
#define CDIM 1024
#define TSEQ 1024
#define NBATCH 2
#define NHEAD 16
#define DHEAD 64
#define FFDIM 4096
#define NROWS 2048            // B*T

// ---------------------------------------------------------------------------
// Embedding: x[row, :] = tok_emb[idx[row]] + pos_emb[row % T]
// ---------------------------------------------------------------------------
__global__ __launch_bounds__(256) void embed_kernel(
    const int* __restrict__ idx, const float* __restrict__ tok,
    const float* __restrict__ pos, float* __restrict__ x)
{
    int row = blockIdx.x;
    int t = row & (TSEQ - 1);
    int token = idx[row];
    int c = threadIdx.x * 4;
    f32x4 te = *(const f32x4*)(tok + (size_t)token * CDIM + c);
    f32x4 pe = *(const f32x4*)(pos + (size_t)t * CDIM + c);
    f32x4 o;
    o.x = te.x + pe.x; o.y = te.y + pe.y; o.z = te.z + pe.z; o.w = te.w + pe.w;
    *(f32x4*)(x + (size_t)row * CDIM + c) = o;
}

// ---------------------------------------------------------------------------
// LayerNorm (one block per row), fp32 in -> bf16 out
// ---------------------------------------------------------------------------
__global__ __launch_bounds__(256) void layernorm_bf16(
    const float* __restrict__ x, const float* __restrict__ s,
    const float* __restrict__ b, unsigned short* __restrict__ out)
{
    __shared__ float r1[256];
    __shared__ float r2[256];
    int row = blockIdx.x;
    int tid = threadIdx.x;
    int c = tid * 4;
    f32x4 v = *(const f32x4*)(x + (size_t)row * CDIM + c);
    float sum = v.x + v.y + v.z + v.w;
    float sq  = v.x * v.x + v.y * v.y + v.z * v.z + v.w * v.w;
    r1[tid] = sum; r2[tid] = sq;
    __syncthreads();
#pragma unroll
    for (int off = 128; off > 0; off >>= 1) {
        if (tid < off) { r1[tid] += r1[tid + off]; r2[tid] += r2[tid + off]; }
        __syncthreads();
    }
    float mean = r1[0] * (1.0f / CDIM);
    float var  = r2[0] * (1.0f / CDIM) - mean * mean;
    float rstd = rsqrtf(var + 1e-5f);
    f32x4 sc = *(const f32x4*)(s + c);
    f32x4 bb = *(const f32x4*)(b + c);
    unsigned short* o = out + (size_t)row * CDIM + c;
    o[0] = f2bf((v.x - mean) * rstd * sc.x + bb.x);
    o[1] = f2bf((v.y - mean) * rstd * sc.y + bb.y);
    o[2] = f2bf((v.z - mean) * rstd * sc.z + bb.z);
    o[3] = f2bf((v.w - mean) * rstd * sc.w + bb.w);
}

// ---------------------------------------------------------------------------
// WMMA GEMM:  C[M,N] = act(A[M,K](bf16) @ W + bias)
//   BT=false: W fp32 [K,N] row-major.   BT=true: W fp32 [N,K] (tied head).
//   STORE: 0 = bf16 store, 1 = fp32 residual add (+=), 2 = fp32 store (stride Nt)
//   ACT:   0 = none, 1 = exact GELU
// Block: 256 thr = 8 waves; tile 128(M) x 128(N), BK=64.
// Waves 4(M) x 2(N); each wave 32x64 = 2x4 WMMA frags x 2 k-steps = 16 WMMA/stage.
// A tile (bf16) moved by the TDM; B tile converted fp32->bf16 through VGPRs.
// ---------------------------------------------------------------------------
template <int ACT, int STORE, bool BT>
__global__ __launch_bounds__(256) void gemm_bf16(
    const unsigned short* __restrict__ A, const float* __restrict__ W,
    const float* __restrict__ bias, float* __restrict__ outF,
    unsigned short* __restrict__ outB, int M, int N, int K, int Nt)
{
    __shared__ __align__(16) unsigned short lds_a[128 * 64];  // [m][k]
    __shared__ __align__(16) unsigned short lds_b[128 * 64];  // [n][k] (transposed)

    int tid  = threadIdx.x;
    int wave = tid >> 5;
    int lane = tid & 31;
    int half = lane >> 4;
    int l16  = lane & 15;
    int wM = (wave >> 1) * 32;
    int wN = (wave & 1) * 64;
    int bm0 = blockIdx.y * 128;
    int bn0 = blockIdx.x * 128;

    unsigned ldsA0 = lds_off(&lds_a[0]);

    v8f acc[2][4];
#pragma unroll
    for (int i = 0; i < 2; ++i)
#pragma unroll
        for (int j = 0; j < 4; ++j) acc[i][j] = zero8();

    for (int k0 = 0; k0 < K; k0 += 64) {
        __syncthreads();
        // ---- stage A tile (128 rows x 64 k, bf16) via Tensor Data Mover ----
        if (wave == 0)
            tdm_load_2d(ldsA0, A + (size_t)bm0 * K + k0,
                        /*tensor_dim0=*/(unsigned)K, /*tensor_dim1=*/128u,
                        /*tile_dim0=*/64u, /*tile_dim1=*/128u,
                        /*stride0=*/(unsigned)K);
        // ---- stage B tile (64k x 128n) as [n][k], fp32 -> bf16 ----
        if (!BT) {
            if (k0 + 64 < K)
                __builtin_prefetch(W + (size_t)(k0 + 64 + (tid >> 5)) * N + bn0 +
                                   (tid & 31) * 4, 0, 0);
#pragma unroll
            for (int j = 0; j < 8; ++j) {
                int cid = j * 256 + tid;             // 0..2047
                int kr = cid >> 5, nq = cid & 31;    // 64 k-rows x 32 n-quads
                f32x4 f = *(const f32x4*)(W + (size_t)(k0 + kr) * N + bn0 + nq * 4);
                int o = (nq * 4) * 64 + kr;
                lds_b[o]       = f2bf(f.x);
                lds_b[o + 64]  = f2bf(f.y);
                lds_b[o + 128] = f2bf(f.z);
                lds_b[o + 192] = f2bf(f.w);
            }
        } else {
#pragma unroll
            for (int j = 0; j < 8; ++j) {
                int cid = j * 256 + tid;            // 0..2047
                int nr = cid >> 4, kq = cid & 15;   // 128 n-rows x 16 k-quads
                int ng = bn0 + nr;
                f32x4 f = fzero4();
                if (ng < Nt) f = *(const f32x4*)(W + (size_t)ng * K + k0 + kq * 4);
                int o = nr * 64 + kq * 4;
                lds_b[o + 0] = f2bf(f.x);
                lds_b[o + 1] = f2bf(f.y);
                lds_b[o + 2] = f2bf(f.z);
                lds_b[o + 3] = f2bf(f.w);
            }
        }
        if (wave == 0) __builtin_amdgcn_s_wait_tensorcnt(0);
        __syncthreads();
        // ---- fragments + WMMA: 2 k-steps of 32 ----
#pragma unroll
        for (int ks = 0; ks < 2; ++ks) {
            Frag af[2], bf[4];
#pragma unroll
            for (int mi = 0; mi < 2; ++mi) {
                int m = wM + mi * 16 + l16;
                af[mi].q[0] = *(const u32x4*)&lds_a[m * 64 + ks * 32 + half * 8];
                af[mi].q[1] = *(const u32x4*)&lds_a[m * 64 + ks * 32 + 16 + half * 8];
            }
#pragma unroll
            for (int ni = 0; ni < 4; ++ni) {
                int n = wN + ni * 16 + l16;
                bf[ni].q[0] = *(const u32x4*)&lds_b[n * 64 + ks * 32 + half * 8];
                bf[ni].q[1] = *(const u32x4*)&lds_b[n * 64 + ks * 32 + 16 + half * 8];
            }
#pragma unroll
            for (int mi = 0; mi < 2; ++mi)
#pragma unroll
                for (int ni = 0; ni < 4; ++ni)
                    acc[mi][ni] = __builtin_amdgcn_wmma_f32_16x16x32_bf16(
                        false, af[mi].v, false, bf[ni].v, (short)0, acc[mi][ni],
                        false, false);
        }
    }

    // ---- epilogue ----
#pragma unroll
    for (int mi = 0; mi < 2; ++mi) {
#pragma unroll
        for (int ni = 0; ni < 4; ++ni) {
            int n = bn0 + wN + ni * 16 + l16;
#pragma unroll
            for (int i = 0; i < 8; ++i) {
                int m = bm0 + wM + mi * 16 + half * 8 + i;
                float v = acc[mi][ni][i];
                if (bias) v += bias[n];
                if (ACT == 1) v = 0.5f * v * (1.0f + erff(v * 0.70710678118654752f));
                if (STORE == 0) {
                    outB[(size_t)m * N + n] = f2bf(v);
                } else if (STORE == 1) {
                    outF[(size_t)m * N + n] += v;
                } else {
                    if (n < Nt) outF[(size_t)m * Nt + n] = v;
                }
            }
        }
    }
}

// ---------------------------------------------------------------------------
// Fused causal flash attention, bf16 WMMA.
// qkv: bf16 [B*T, 3C]; out: bf16 [B*T, C].
// grid = (T/128, B*H); 8 waves/block, one 16-row query tile per wave.
// K tile staged via async global->LDS; V staged transposed through VGPRs.
// ---------------------------------------------------------------------------
__global__ __launch_bounds__(256) void attn_kernel(
    const unsigned short* __restrict__ qkv, unsigned short* __restrict__ out)
{
    __shared__ __align__(16) unsigned short lds_k[32 * 64];   // [key][d]
    __shared__ __align__(16) unsigned short lds_vT[64 * 32];  // [d][key]
    __shared__ __align__(16) unsigned short lds_p[8][16 * 32];// per-wave P

    const float scale = 0.125f;  // 1/sqrt(64)
    int bh = blockIdx.y;
    int b  = bh >> 4;
    int h  = bh & 15;
    int qb0 = blockIdx.x * 128;
    int tid  = threadIdx.x;
    int wave = tid >> 5;
    int lane = tid & 31;
    int half = lane >> 4;
    int l16  = lane & 15;
    int qrow0 = qb0 + wave * 16;

    const unsigned short* base = qkv + ((size_t)b * TSEQ) * (3 * CDIM) + h * DHEAD;
    const unsigned short* kbase = base + CDIM;      // uniform per block
    unsigned ldsK0 = lds_off(&lds_k[0]);

    // Q fragments (16 rows x 64 dims -> two k-steps of 32)
    Frag aq[2];
#pragma unroll
    for (int ks = 0; ks < 2; ++ks) {
        const unsigned short* p = base + (size_t)(qrow0 + l16) * (3 * CDIM) + ks * 32;
        aq[ks].q[0] = *(const u32x4*)(p + half * 8);
        aq[ks].q[1] = *(const u32x4*)(p + 16 + half * 8);
    }

    v8f o[4];
#pragma unroll
    for (int i = 0; i < 4; ++i) o[i] = zero8();
    float mrow[8], lrow[8];
#pragma unroll
    for (int i = 0; i < 8; ++i) { mrow[i] = -3.0e38f; lrow[i] = 0.0f; }

    int jmax_me  = (qrow0 + 15) >> 5;
    int jmax_blk = (qb0 + 127) >> 5;

    for (int j = 0; j <= jmax_blk; ++j) {
        __syncthreads();
        {   // stage 32 keys x 64 dims: K async (row-major), V via VGPRs (transposed)
            int row = tid >> 3, c8 = tid & 7;
            unsigned goff = (unsigned)((((j * 32 + row) * (3 * CDIM)) + c8 * 8) * 2);
            async_b128(ldsK0 + (unsigned)((row * 64 + c8 * 8) * 2), goff, kbase);
            const unsigned short* vp =
                base + 2 * CDIM + (size_t)(j * 32 + row) * (3 * CDIM) + c8 * 8;
            u32x4 vv = *(const u32x4*)vp;
            unsigned short tmp[8];
            *(u32x4*)tmp = vv;
#pragma unroll
            for (int i = 0; i < 8; ++i) lds_vT[(c8 * 8 + i) * 32 + row] = tmp[i];
        }
        wait_async0();
        __syncthreads();
        if (j > jmax_me) continue;

        // scores S (16 rows x 32 keys) as two 16x16 C-fragments
        v8f sfr[2];
        sfr[0] = zero8(); sfr[1] = zero8();
#pragma unroll
        for (int sub = 0; sub < 2; ++sub) {
            int key = sub * 16 + l16;
#pragma unroll
            for (int ks = 0; ks < 2; ++ks) {
                Frag bk;
                bk.q[0] = *(const u32x4*)&lds_k[key * 64 + ks * 32 + half * 8];
                bk.q[1] = *(const u32x4*)&lds_k[key * 64 + ks * 32 + 16 + half * 8];
                sfr[sub] = __builtin_amdgcn_wmma_f32_16x16x32_bf16(
                    false, aq[ks].v, false, bk.v, (short)0, sfr[sub], false, false);
            }
        }
        // online softmax (rows live in 16-lane halves of the wave32 C layout)
#pragma unroll
        for (int i = 0; i < 8; ++i) {
            int qr = qrow0 + half * 8 + i;
            float s0 = sfr[0][i] * scale;
            float s1 = sfr[1][i] * scale;
            if (j * 32 + l16 > qr)      s0 = -3.0e38f;
            if (j * 32 + 16 + l16 > qr) s1 = -3.0e38f;
            float mx = fmaxf(s0, s1);
            mx = fmaxf(mx, __shfl_xor(mx, 1, 32));
            mx = fmaxf(mx, __shfl_xor(mx, 2, 32));
            mx = fmaxf(mx, __shfl_xor(mx, 4, 32));
            mx = fmaxf(mx, __shfl_xor(mx, 8, 32));
            float nm = fmaxf(mrow[i], mx);
            float f  = __expf(mrow[i] - nm);
            float p0 = __expf(s0 - nm);
            float p1 = __expf(s1 - nm);
            float rs = p0 + p1;
            rs += __shfl_xor(rs, 1, 32);
            rs += __shfl_xor(rs, 2, 32);
            rs += __shfl_xor(rs, 4, 32);
            rs += __shfl_xor(rs, 8, 32);
            lrow[i] = lrow[i] * f + rs;
            mrow[i] = nm;
#pragma unroll
            for (int ni = 0; ni < 4; ++ni) o[ni][i] *= f;
            int ml = half * 8 + i;
            lds_p[wave][ml * 32 + l16]      = f2bf(p0);
            lds_p[wave][ml * 32 + 16 + l16] = f2bf(p1);
        }
        // P (C-layout) -> A-layout via per-wave LDS, then O += P @ V
        Frag ap;
        ap.q[0] = *(const u32x4*)&lds_p[wave][l16 * 32 + half * 8];
        ap.q[1] = *(const u32x4*)&lds_p[wave][l16 * 32 + 16 + half * 8];
#pragma unroll
        for (int ni = 0; ni < 4; ++ni) {
            Frag bv;
            bv.q[0] = *(const u32x4*)&lds_vT[(ni * 16 + l16) * 32 + half * 8];
            bv.q[1] = *(const u32x4*)&lds_vT[(ni * 16 + l16) * 32 + 16 + half * 8];
            o[ni] = __builtin_amdgcn_wmma_f32_16x16x32_bf16(
                false, ap.v, false, bv.v, (short)0, o[ni], false, false);
        }
    }

    // epilogue: normalize and store bf16
#pragma unroll
    for (int i = 0; i < 8; ++i) {
        float inv = 1.0f / lrow[i];
        int m = qrow0 + half * 8 + i;
        unsigned short* op = out + ((size_t)(b * TSEQ + m)) * CDIM + h * DHEAD;
#pragma unroll
        for (int ni = 0; ni < 4; ++ni)
            op[ni * 16 + l16] = f2bf(o[ni][i] * inv);
    }
}

// ---------------------------------------------------------------------------
// Host orchestration
// ---------------------------------------------------------------------------
extern "C" void kernel_launch(void* const* d_in, const int* in_sizes, int n_in,
                              void* d_out, int out_size, void* d_ws, size_t ws_size,
                              hipStream_t stream)
{
    const int*   idx     = (const int*)  d_in[0];
    const float* tok_emb = (const float*)d_in[1];
    const float* pos_emb = (const float*)d_in[2];
    const float* qkv_w   = (const float*)d_in[3];
    const float* qkv_b   = (const float*)d_in[4];
    const float* out_w   = (const float*)d_in[5];
    const float* out_b   = (const float*)d_in[6];
    const float* ln1_s   = (const float*)d_in[7];
    const float* ln1_b   = (const float*)d_in[8];
    const float* ln2_s   = (const float*)d_in[9];
    const float* ln2_b   = (const float*)d_in[10];
    const float* ffn_w1  = (const float*)d_in[11];
    const float* ffn_b1  = (const float*)d_in[12];
    const float* ffn_w2  = (const float*)d_in[13];
    const float* ffn_b2  = (const float*)d_in[14];
    const float* lnf_s   = (const float*)d_in[15];
    const float* lnf_b   = (const float*)d_in[16];

    // Workspace layout (all 16B aligned); total ~44 MB
    char* ws = (char*)d_ws;
    float*          x      = (float*)(ws + 0);                       //  8 MB fp32 residual
    unsigned short* hbf    = (unsigned short*)(ws + 8388608);        //  4 MB LN out (bf16)
    unsigned short* qkvbf  = (unsigned short*)(ws + 12582912);       // 12 MB qkv (bf16)
    unsigned short* attnbf = (unsigned short*)(ws + 25165824);       //  4 MB attn out (bf16)
    unsigned short* ffnbf  = (unsigned short*)(ws + 29360128);       // 16 MB ffn hidden (bf16)

    const int V = 50257;

    embed_kernel<<<NROWS, 256, 0, stream>>>(idx, tok_emb, pos_emb, x);

    for (int l = 0; l < 8; ++l) {
        layernorm_bf16<<<NROWS, 256, 0, stream>>>(
            x, ln1_s + l * CDIM, ln1_b + l * CDIM, hbf);
        gemm_bf16<0, 0, false><<<dim3(3 * CDIM / 128, NROWS / 128), 256, 0, stream>>>(
            hbf, qkv_w + (size_t)l * CDIM * 3 * CDIM, qkv_b + (size_t)l * 3 * CDIM,
            nullptr, qkvbf, NROWS, 3 * CDIM, CDIM, 3 * CDIM);
        attn_kernel<<<dim3(TSEQ / 128, NBATCH * NHEAD), 256, 0, stream>>>(
            qkvbf, attnbf);
        gemm_bf16<0, 1, false><<<dim3(CDIM / 128, NROWS / 128), 256, 0, stream>>>(
            attnbf, out_w + (size_t)l * CDIM * CDIM, out_b + (size_t)l * CDIM,
            x, nullptr, NROWS, CDIM, CDIM, CDIM);
        layernorm_bf16<<<NROWS, 256, 0, stream>>>(
            x, ln2_s + l * CDIM, ln2_b + l * CDIM, hbf);
        gemm_bf16<1, 0, false><<<dim3(FFDIM / 128, NROWS / 128), 256, 0, stream>>>(
            hbf, ffn_w1 + (size_t)l * CDIM * FFDIM, ffn_b1 + (size_t)l * FFDIM,
            nullptr, ffnbf, NROWS, FFDIM, CDIM, FFDIM);
        gemm_bf16<0, 1, false><<<dim3(CDIM / 128, NROWS / 128), 256, 0, stream>>>(
            ffnbf, ffn_w2 + (size_t)l * FFDIM * CDIM, ffn_b2 + (size_t)l * CDIM,
            x, nullptr, NROWS, CDIM, FFDIM, CDIM);
    }

    layernorm_bf16<<<NROWS, 256, 0, stream>>>(x, lnf_s, lnf_b, hbf);
    // tied head: logits = h @ tok_emb^T   (W given as [V, C] -> BT path)
    gemm_bf16<0, 2, true><<<dim3((V + 127) / 128, NROWS / 128), 256, 0, stream>>>(
        hbf, tok_emb, nullptr, (float*)d_out, nullptr, NROWS, V, CDIM, V);
}
